// RNNBase_7516192768859
// MI455X (gfx1250) — matrix operand order repaired and checked by reading, not compiled, not run on hardware
//
#include <hip/hip_runtime.h>
#include <hip/hip_bf16.h>
#include <stdint.h>

// ---------------------------------------------------------------------------
// Problem constants (from reference)
// ---------------------------------------------------------------------------
#define SEQ_LEN 128
#define BATCH   64
#define VOCAB   10000
#define HID     1024
#define NLAYERS 2

typedef __attribute__((ext_vector_type(16))) __bf16 v16bf;
typedef __attribute__((ext_vector_type(8)))  float  v8f;

union Frag16 {           // 16 bf16 = 8 VGPRs, loaded as two 16-byte vectors
    v16bf v;
    uint4 q[2];
};

__device__ __forceinline__ uint4 ld16(const __bf16* p) {
    return *reinterpret_cast<const uint4*>(p);
}

__device__ __forceinline__ v8f wmma_bf16(const Frag16& a, const Frag16& b, v8f c) {
    // D(f32 16x16) = A(bf16 16x32) * B(bf16 32x16) + C
    return __builtin_amdgcn_wmma_f32_16x16x32_bf16(
        /*neg_a=*/false, a.v, /*neg_b=*/false, b.v,
        /*c_mod=*/(short)0, c, /*reuse_a=*/false, /*reuse_b=*/false);
}

// Branch-free tanh for the recurrence critical path.
__device__ __forceinline__ float fast_tanh(float x) {
#if __has_builtin(__builtin_amdgcn_tanhf)
    return __builtin_amdgcn_tanhf(x);
#else
    x = fminf(fmaxf(x, -15.0f), 15.0f);
    const float e = __expf(2.0f * x);
    return (e - 1.0f) * __builtin_amdgcn_rcpf(e + 1.0f);
#endif
}

// ---------------------------------------------------------------------------
// Embedding gather + f32 -> bf16 convert.
// ---------------------------------------------------------------------------
__global__ void __launch_bounds__(256)
embed_gather(const int* __restrict__ inputs,
             const float* __restrict__ emb,
             __bf16* __restrict__ xseq) {
    const int row = blockIdx.x;                 // 0 .. SEQ_LEN*BATCH-1
    const int tok = inputs[row];
    const float*  e = emb  + (size_t)tok * HID;
    __bf16*       o = xseq + (size_t)row * HID;
    for (int i = threadIdx.x; i < HID; i += blockDim.x)
        o[i] = (__bf16)e[i];
}

// ---------------------------------------------------------------------------
// LDS-tiled transpose with bf16 convert: dst[c][r] = bf16(src[r][c])
// block = (32,8), grid = (ceil(C/32), ceil(R/32), Z)
// ---------------------------------------------------------------------------
__global__ void __launch_bounds__(256)
transpose_to_bf16(const float* __restrict__ src,
                  __bf16* __restrict__ dst,
                  int R, int C, size_t zstride) {
    __shared__ float tile[32][33];
    src += (size_t)blockIdx.z * zstride;
    dst += (size_t)blockIdx.z * zstride;
    const int cb = blockIdx.x * 32, rb = blockIdx.y * 32;
    const int tx = threadIdx.x, ty = threadIdx.y;
#pragma unroll
    for (int i = 0; i < 32; i += 8) {
        int r = rb + ty + i, c = cb + tx;
        tile[ty + i][tx] = (r < R && c < C) ? src[(size_t)r * C + c] : 0.0f;
    }
    __syncthreads();
#pragma unroll
    for (int i = 0; i < 32; i += 8) {
        int c = cb + ty + i, r = rb + tx;
        if (c < C && r < R)
            dst[(size_t)c * R + r] = (__bf16)tile[tx][ty + i];
    }
}

// ---------------------------------------------------------------------------
// Initial hidden -> bf16 state buffers; resets grid-barrier counter so every
// kernel_launch / graph replay is deterministic.
// ---------------------------------------------------------------------------
__global__ void __launch_bounds__(256)
hid_init(const float* __restrict__ hidden,
         __bf16* __restrict__ h1a,
         __bf16* __restrict__ h2a,
         unsigned* __restrict__ bar) {
    const int i = blockIdx.x * blockDim.x + threadIdx.x;
    if (i == 0) bar[0] = 0u;
    if (i < BATCH * HID) {
        h1a[i] = (__bf16)hidden[i];
        h2a[i] = (__bf16)hidden[BATCH * HID + i];
    }
}

// ---------------------------------------------------------------------------
// Software grid barrier (32 blocks, all resident)
// ---------------------------------------------------------------------------
__device__ __forceinline__ void grid_sync(unsigned* bar, unsigned target) {
    __syncthreads();
    if (threadIdx.x == 0) {
        __threadfence();
        __hip_atomic_fetch_add(bar, 1u, __ATOMIC_ACQ_REL, __HIP_MEMORY_SCOPE_AGENT);
        while (__hip_atomic_load(bar, __ATOMIC_ACQUIRE, __HIP_MEMORY_SCOPE_AGENT) < target)
            __builtin_amdgcn_s_sleep(1);
    }
    __syncthreads();
}

// ---------------------------------------------------------------------------
// One RNN layer K-loop for one 16x16 tile, software-pipelined with two
// explicit fragment buffer sets so loads for chunk k+32 stay in flight while
// chunk k's WMMAs execute (partial s_wait_loadcnt instead of full drains).
//   acc += X[m0:,:] * WxT^T + H[m0:,:] * WhT^T   (K = HID, chunks of 32)
// ---------------------------------------------------------------------------
__device__ __forceinline__ void rnn_layer_mac(const __bf16* __restrict__ Xb,
                                              const __bf16* __restrict__ Hb,
                                              const __bf16* __restrict__ BxT,
                                              const __bf16* __restrict__ BhT,
                                              int mrow, int nglob, int half,
                                              v8f& acc) {
    // Fold lane-half offsets into the bases: A uses +half*16, B uses +half*16.
    const __bf16* xp  = Xb  + (size_t)mrow  * HID + half * 16;
    const __bf16* hp  = Hb  + (size_t)mrow  * HID + half * 16;
    const __bf16* bxp = BxT + (size_t)nglob * HID + half * 16;
    const __bf16* bhp = BhT + (size_t)nglob * HID + half * 16;

    Frag16 xa0, bx0, ha0, bh0, xa1, bx1, ha1, bh1;

    // prologue: chunk 0 -> buffer set 0
    xa0.q[0] = ld16(xp);       xa0.q[1] = ld16(xp + 8);
    bx0.q[0] = ld16(bxp);      bx0.q[1] = ld16(bxp + 8);
    ha0.q[0] = ld16(hp);       ha0.q[1] = ld16(hp + 8);
    bh0.q[0] = ld16(bhp);      bh0.q[1] = ld16(bhp + 8);

#pragma unroll 2
    for (int kc = 0; kc < HID; kc += 64) {
        const int k1 = kc + 32;                 // always < HID
        const int k2 = (kc + 64) & (HID - 1);   // wraps to 0 on last iter (redundant, harmless)

        // load chunk k1 -> buffer set 1 (in flight during buf0's WMMAs)
        xa1.q[0] = ld16(xp + k1);       xa1.q[1] = ld16(xp + k1 + 8);
        bx1.q[0] = ld16(bxp + k1);      bx1.q[1] = ld16(bxp + k1 + 8);
        ha1.q[0] = ld16(hp + k1);       ha1.q[1] = ld16(hp + k1 + 8);
        bh1.q[0] = ld16(bhp + k1);      bh1.q[1] = ld16(bhp + k1 + 8);

        acc = wmma_bf16(xa0, bx0, acc);
        acc = wmma_bf16(ha0, bh0, acc);

        // load chunk k2 -> buffer set 0 (in flight during buf1's WMMAs)
        xa0.q[0] = ld16(xp + k2);       xa0.q[1] = ld16(xp + k2 + 8);
        bx0.q[0] = ld16(bxp + k2);      bx0.q[1] = ld16(bxp + k2 + 8);
        ha0.q[0] = ld16(hp + k2);       ha0.q[1] = ld16(hp + k2 + 8);
        bh0.q[0] = ld16(bhp + k2);      bh0.q[1] = ld16(bhp + k2 + 8);

        acc = wmma_bf16(xa1, bx1, acc);
        acc = wmma_bf16(ha1, bh1, acc);
    }
}

// ---------------------------------------------------------------------------
// One full timestep (both layers + 2 grid barriers) with FIXED in/out buffer
// pointers (no pointer PHI -> all loads stay GLOBAL, not FLAT).
// ---------------------------------------------------------------------------
__device__ __forceinline__ void rnn_step(int t,
                                         const __bf16* __restrict__ xseq,
                                         const __bf16* __restrict__ WxT0,
                                         const __bf16* __restrict__ WhT0,
                                         const __bf16* __restrict__ WxT1,
                                         const __bf16* __restrict__ WhT1,
                                         float bn0, float bn1,
                                         const __bf16* __restrict__ h1_in,
                                         __bf16* __restrict__ h1_out,
                                         const __bf16* __restrict__ h2_in,
                                         __bf16* __restrict__ h2_out,
                                         __bf16* __restrict__ topseq,
                                         float* __restrict__ hidden_out,
                                         unsigned* __restrict__ bar,
                                         unsigned& target, unsigned nb,
                                         int m0, int mrow, int nglob, int half) {
    // ---- layer 0: h1 = tanh(x_t*Wx0 + h1_prev*Wh0 + b0) ----
    {
        const __bf16* Xb = xseq + (size_t)t * BATCH * HID;
        v8f acc = {};
        rnn_layer_mac(Xb, h1_in, WxT0, WhT0, mrow, nglob, half, acc);
#pragma unroll
        for (int r = 0; r < 8; ++r) {
            const float v = fast_tanh(acc[r] + bn0);
            const int   m = m0 + r + half * 8;
            h1_out[(size_t)m * HID + nglob] = (__bf16)v;
            if (t == SEQ_LEN - 1)
                hidden_out[(size_t)m * HID + nglob] = v;
        }
    }
    grid_sync(bar, target += nb);

    // ---- layer 1: h2 = tanh(h1*Wx1 + h2_prev*Wh1 + b1) ----
    {
        v8f acc = {};
        rnn_layer_mac(h1_out, h2_in, WxT1, WhT1, mrow, nglob, half, acc);
        __bf16* Top = topseq + (size_t)t * BATCH * HID;
#pragma unroll
        for (int r = 0; r < 8; ++r) {
            const float v = fast_tanh(acc[r] + bn1);
            const int   m = m0 + r + half * 8;
            h2_out[(size_t)m * HID + nglob] = (__bf16)v;
            Top[(size_t)m * HID + nglob]    = (__bf16)v;
            if (t == SEQ_LEN - 1)
                hidden_out[(size_t)(BATCH * HID) + m * HID + nglob] = v;
        }
    }
    grid_sync(bar, target += nb);
}

// ---------------------------------------------------------------------------
// Persistent recurrent kernel: 32 blocks x 256 threads = 256 waves,
// one wave per 16x16 tile of the [64 x 1024] hidden state.
// Time loop unrolled by 2 (a->b then b->a) so buffers are compile-time fixed.
// ---------------------------------------------------------------------------
__global__ void __launch_bounds__(256)
rnn_persistent(const __bf16* __restrict__ xseq,
               const __bf16* __restrict__ WxT,  // [L][N][K] bf16
               const __bf16* __restrict__ WhT,  // [L][N][K] bf16
               const float*  __restrict__ bias, // [L][HID]
               __bf16* __restrict__ h1a, __bf16* __restrict__ h1b,
               __bf16* __restrict__ h2a, __bf16* __restrict__ h2b,
               __bf16* __restrict__ topseq,     // [S][B][HID] bf16
               float*  __restrict__ hidden_out, // [L][B][HID] f32
               unsigned* __restrict__ bar) {
    const int lane  = threadIdx.x & 31;
    const int wave  = (blockIdx.x * blockDim.x + threadIdx.x) >> 5;  // 0..255
    const int mt    = wave >> 6;            // 0..3   (BATCH/16)
    const int nt    = wave & 63;            // 0..63  (HID/16)
    const int m0    = mt * 16;
    const int half  = lane >> 4;            // 0: lanes 0-15, 1: lanes 16-31
    const int mrow  = m0 + (lane & 15);     // A-matrix row for this lane
    const int nglob = nt * 16 + (lane & 15);// C/D column for this lane

    const __bf16* WxT0 = WxT;  const __bf16* WxT1 = WxT + (size_t)HID * HID;
    const __bf16* WhT0 = WhT;  const __bf16* WhT1 = WhT + (size_t)HID * HID;
    const float bn0 = bias[nglob];
    const float bn1 = bias[HID + nglob];

    unsigned target = 0;
    const unsigned nb = gridDim.x;

    for (int t = 0; t < SEQ_LEN; t += 2) {
        // even step: read a-buffers, write b-buffers
        rnn_step(t,     xseq, WxT0, WhT0, WxT1, WhT1, bn0, bn1,
                 h1a, h1b, h2a, h2b, topseq, hidden_out,
                 bar, target, nb, m0, mrow, nglob, half);
        // odd step: read b-buffers, write a-buffers
        rnn_step(t + 1, xseq, WxT0, WhT0, WxT1, WhT1, bn0, bn1,
                 h1b, h1a, h2b, h2a, topseq, hidden_out,
                 bar, target, nb, m0, mrow, nglob, half);
    }
}

// ---------------------------------------------------------------------------
// Output projection: logits[8192 x 10000] = top[8192 x 1024] * Wo + bo.
// One wave computes a 16x80 strip (A-fragment reused across 5 N-tiles),
// software-pipelined the same way (two buffer sets of 1 A + 5 B fragments).
// 10000 = 625 * 16 exactly; 625 = 125 groups of 5 tiles.
// ---------------------------------------------------------------------------
#define OUT_MT (SEQ_LEN * BATCH / 16)   // 512
#define OUT_NG (VOCAB / 80)             // 125

__global__ void __launch_bounds__(256)
out_gemm(const __bf16* __restrict__ top,
         const __bf16* __restrict__ WoT,   // [VOCAB][HID] bf16
         const float*  __restrict__ bo,
         float* __restrict__ logits) {
    const int lane = threadIdx.x & 31;
    const int wv   = (blockIdx.x * blockDim.x + threadIdx.x) >> 5;
    const int mt   = wv / OUT_NG;          // 0..511
    const int g    = wv % OUT_NG;          // 0..124
    const int m0   = mt * 16;
    const int half = lane >> 4;
    const int mrow = m0 + (lane & 15);
    const int nl   = lane & 15;

    v8f acc[5] = {v8f{}, v8f{}, v8f{}, v8f{}, v8f{}};
    const __bf16* Arow = top + (size_t)mrow * HID + half * 16;
    const __bf16* Brow[5];
#pragma unroll
    for (int s = 0; s < 5; ++s)
        Brow[s] = WoT + (size_t)(g * 80 + s * 16 + nl) * HID + half * 16;

    Frag16 a0, a1, b0[5], b1[5];
    // prologue: chunk 0 -> buffer set 0
    a0.q[0] = ld16(Arow);  a0.q[1] = ld16(Arow + 8);
#pragma unroll
    for (int s = 0; s < 5; ++s) {
        b0[s].q[0] = ld16(Brow[s]);  b0[s].q[1] = ld16(Brow[s] + 8);
    }

    for (int kc = 0; kc < HID; kc += 64) {
        const int k1 = kc + 32;
        const int k2 = (kc + 64) & (HID - 1);

        a1.q[0] = ld16(Arow + k1);  a1.q[1] = ld16(Arow + k1 + 8);
#pragma unroll
        for (int s = 0; s < 5; ++s) {
            b1[s].q[0] = ld16(Brow[s] + k1);  b1[s].q[1] = ld16(Brow[s] + k1 + 8);
        }
#pragma unroll
        for (int s = 0; s < 5; ++s)
            acc[s] = wmma_bf16(a0, b0[s], acc[s]);

        a0.q[0] = ld16(Arow + k2);  a0.q[1] = ld16(Arow + k2 + 8);
#pragma unroll
        for (int s = 0; s < 5; ++s) {
            b0[s].q[0] = ld16(Brow[s] + k2);  b0[s].q[1] = ld16(Brow[s] + k2 + 8);
        }
#pragma unroll
        for (int s = 0; s < 5; ++s)
            acc[s] = wmma_bf16(a1, b1[s], acc[s]);
    }

#pragma unroll
    for (int s = 0; s < 5; ++s) {
        const int v  = g * 80 + s * 16 + nl;
        const float bias = bo[v];
#pragma unroll
        for (int r = 0; r < 8; ++r) {
            const int m = m0 + r + half * 8;
            logits[(size_t)m * VOCAB + v] = acc[s][r] + bias;
        }
    }
}

// ---------------------------------------------------------------------------
// kernel_launch
// ---------------------------------------------------------------------------
extern "C" void kernel_launch(void* const* d_in, const int* in_sizes, int n_in,
                              void* d_out, int out_size, void* d_ws, size_t ws_size,
                              hipStream_t stream) {
    const int*   inputs = (const int*)  d_in[0];  // [128,64]
    const float* hidden = (const float*)d_in[1];  // [2,64,1024]
    const float* emb    = (const float*)d_in[2];  // [10000,1024]
    const float* Wx     = (const float*)d_in[3];  // [2,1024,1024]
    const float* Wh     = (const float*)d_in[4];  // [2,1024,1024]
    const float* b      = (const float*)d_in[5];  // [2,1024]
    const float* Wo     = (const float*)d_in[6];  // [1024,10000]
    const float* bo     = (const float*)d_in[7];  // [10000]

    float* logits     = (float*)d_out;                                   // [128,64,10000]
    float* hidden_out = (float*)d_out + (size_t)SEQ_LEN * BATCH * VOCAB; // [2,64,1024]

    // workspace carve-up (all offsets 256-aligned); total ~63 MB
    char* p = (char*)d_ws;
    auto carve = [&](size_t bytes) { char* r = p; p += (bytes + 255) & ~(size_t)255; return r; };
    __bf16* xseq = (__bf16*)carve((size_t)SEQ_LEN * BATCH * HID * 2);
    __bf16* WxT  = (__bf16*)carve((size_t)NLAYERS * HID * HID * 2);
    __bf16* WhT  = (__bf16*)carve((size_t)NLAYERS * HID * HID * 2);
    __bf16* WoT  = (__bf16*)carve((size_t)VOCAB * HID * 2);
    __bf16* h1a  = (__bf16*)carve((size_t)BATCH * HID * 2);
    __bf16* h1b  = (__bf16*)carve((size_t)BATCH * HID * 2);
    __bf16* h2a  = (__bf16*)carve((size_t)BATCH * HID * 2);
    __bf16* h2b  = (__bf16*)carve((size_t)BATCH * HID * 2);
    __bf16* topq = (__bf16*)carve((size_t)SEQ_LEN * BATCH * HID * 2);
    unsigned* bar = (unsigned*)carve(256);

    // 1) embedding gather -> bf16
    embed_gather<<<SEQ_LEN * BATCH, 256, 0, stream>>>(inputs, emb, xseq);

    // 2) weight transpose+convert: Wx/Wh [K][N] -> [N][K] bf16, per layer (z)
    {
        dim3 blk(32, 8, 1);
        dim3 grd(HID / 32, HID / 32, NLAYERS);
        transpose_to_bf16<<<grd, blk, 0, stream>>>(Wx, WxT, HID, HID, (size_t)HID * HID);
        transpose_to_bf16<<<grd, blk, 0, stream>>>(Wh, WhT, HID, HID, (size_t)HID * HID);
        dim3 grdo((VOCAB + 31) / 32, HID / 32, 1);
        transpose_to_bf16<<<grdo, blk, 0, stream>>>(Wo, WoT, HID, VOCAB, 0);
    }

    // 3) initial hidden -> bf16 state, reset barrier counter
    hid_init<<<(BATCH * HID + 255) / 256, 256, 0, stream>>>(hidden, h1a, h2a, bar);

    // 4) persistent recurrent kernel: 32 blocks x 8 waves = 256 tile-waves
    rnn_persistent<<<32, 256, 0, stream>>>(xseq, WxT, WhT, b,
                                           h1a, h1b, h2a, h2b,
                                           topq, hidden_out, bar);

    // 5) output projection: 512 * 125 waves, 8 waves/block
    out_gemm<<<OUT_MT * OUT_NG / 8, 256, 0, stream>>>(topq, WoT, bo, logits);
}
